// WideAndDeepModel_37014028157488
// MI455X (gfx1250) — matrix operand order repaired and checked by reading, not compiled, not run on hardware
//
#include <hip/hip_runtime.h>
#include <hip/hip_bf16.h>
#include <math.h>

typedef __attribute__((ext_vector_type(16))) __bf16 v16bf;
typedef __attribute__((ext_vector_type(8)))  float  v8f;
typedef __attribute__((ext_vector_type(4)))  unsigned int v4u;
typedef __attribute__((ext_vector_type(8)))  int v8i;
typedef __attribute__((ext_vector_type(4)))  int v4i;

#define N_FEAT 256
#define HIDDEN 128
#define N_PAIRS 32640
#define BATCH 4096

#if defined(__has_builtin)
#if __has_builtin(__builtin_amdgcn_tensor_load_to_lds) && __has_builtin(__builtin_amdgcn_s_wait_tensorcnt)
#define HAVE_TDM 1
#endif
#endif
#ifndef HAVE_TDM
#define HAVE_TDM 0
#endif

// Byte offset of a __shared__ object inside the workgroup LDS allocation
// (generic -> addrspace(3) cast, then truncate).
__device__ static inline unsigned lds_byte_offset(const void* p) {
    return (unsigned)(unsigned long long)
        (__attribute__((address_space(3))) const char*)p;
}

// ---------------------------------------------------------------------------
// Prepack kernels: build WMMA-operand-layout bf16 tiles.
// B-operand tile (32x16, KxN), lane L, elem e: K = (L>>4)*16 + e, N = L&15
// A-operand tile (16x32, MxK), lane L, elem e: M = L&15,
//   K = (L>>4)*16 + e
// ---------------------------------------------------------------------------

__global__ void pack_M_kernel(const float* __restrict__ w_wide,
                              const float* __restrict__ Wo,
                              __bf16* __restrict__ dst) {
    int idx = blockIdx.x * blockDim.x + threadIdx.x;   // 65536 threads
    int e    = idx & 15;
    int lane = (idx >> 4) & 31;
    int t    = idx >> 9;
    int ntile = t & 15;
    int ktile = t >> 4;
    int half = lane >> 4;
    int row = ktile * 32 + half * 16 + e;
    int col = ntile * 16 + (lane & 15);
    float val = 0.0f;
    if (row != col) {
        int i = row < col ? row : col;
        int j = row < col ? col : row;
        int p = (i * (2 * N_FEAT - i - 1)) / 2 + (j - i - 1);
        val = 0.5f * w_wide[p] * Wo[p];
    }
    dst[idx] = (__bf16)val;
}

__global__ void pack_W_kernel(const float* __restrict__ src,
                              __bf16* __restrict__ dst, int K, int N) {
    int idx = blockIdx.x * blockDim.x + threadIdx.x;
    if (idx >= K * N) return;
    int e    = idx & 15;
    int lane = (idx >> 4) & 31;
    int t    = idx >> 9;
    int NT   = N >> 4;
    int ntile = t % NT;
    int ktile = t / NT;
    int half = lane >> 4;
    int row = ktile * 32 + half * 16 + e;
    int col = ntile * 16 + (lane & 15);
    dst[idx] = (__bf16)src[row * N + col];
}

__global__ void pack_X_kernel(const float* __restrict__ x,
                              __bf16* __restrict__ dst) {
    int idx = blockIdx.x * blockDim.x + threadIdx.x;   // 1048576 threads
    int e    = idx & 15;
    int lane = (idx >> 4) & 31;
    int t    = idx >> 9;
    int ktile = t & 7;
    int mtile = t >> 3;
    int half = lane >> 4;
    int K = half * 16 + e;
    int row = mtile * 16 + (lane & 15);
    int col = ktile * 32 + K;
    dst[idx] = (__bf16)x[row * N_FEAT + col];
}

// ---------------------------------------------------------------------------
// Main fused kernel: one wave32 per 16-row batch strip.
// TDM pulls the 16x256 f32 x-tile into LDS; three WMMA GEMM phases with
// 2-wide N blocking (independent accumulator chains) and double-buffered
// B-tile register loads; fused dot-product epilogues; sigmoid.
// ---------------------------------------------------------------------------
__global__ __launch_bounds__(32) void wd_main_kernel(
    const float* __restrict__ x,
    const float* __restrict__ b1,
    const float* __restrict__ b2,
    const float* __restrict__ Wo,
    const float* __restrict__ bo,
    const v16bf* __restrict__ Xp,
    const v16bf* __restrict__ Mp,
    const v16bf* __restrict__ W1p,
    const v16bf* __restrict__ W2p,
    float* __restrict__ out) {

    __shared__ float  lds_x[16][N_FEAT];     // 16 KB
    __shared__ __bf16 lds_h1[16][HIDDEN];    // 4 KB
    __shared__ float  lds_red[32][8];        // 1 KB

    const int lane  = threadIdx.x;
    const int mtile = blockIdx.x;
    const int nl    = lane & 15;
    const int half  = lane >> 4;

    // ---- Stage the f32 x strip (tile: 256 cols x 16 rows) into LDS ----
#if HAVE_TDM
    {
        unsigned long long gaddr =
            (unsigned long long)(x + (size_t)mtile * 16 * N_FEAT);
        unsigned ldsa = lds_byte_offset(&lds_x[0][0]);
        // D# group 0: count=1 | lds_addr | global_addr(57b) | type=2
        v4u g0 = { 1u, ldsa, (unsigned)gaddr,
                   (unsigned)((gaddr >> 32) & 0x01FFFFFFu) | (2u << 30) };
        // D# group 1: data_size=4B; tensor 256 x 4096; tile 256 x 16; stride 256
        v8i g1 = { (int)0x00020000,          // workgroup_mask=0, data_size=2
                   (int)(256u << 16),        // tensor_dim0 = 256 (lo16 in [63:48])
                   (int)(4096u << 16),       // tensor_dim0 hi=0, tensor_dim1 lo=4096
                   (int)(256u << 16),        // tensor_dim1 hi=0, tile_dim0 = 256
                   (int)16,                  // tile_dim1 = 16, tile_dim2 = 0
                   (int)256,                 // tensor_dim0_stride = 256
                   0, 0 };
        v4i gz4 = { 0, 0, 0, 0 };
        v8i gz8 = { 0, 0, 0, 0, 0, 0, 0, 0 };
        __builtin_amdgcn_tensor_load_to_lds(g0, g1, gz4, gz4, gz8, 0);
        __builtin_amdgcn_s_wait_tensorcnt(0);
    }
#else
    {
        const float4* src = (const float4*)(x + mtile * 16 * N_FEAT);
        float4* dstl = (float4*)(&lds_x[0][0]);
        #pragma unroll
        for (int i = 0; i < 32; ++i)
            dstl[i * 32 + lane] = src[i * 32 + lane];
    }
#endif

    // A tiles of X: reused by GEMM1 (deep) and GEMM3 (wide).
    v16bf a[8];
    #pragma unroll
    for (int kt = 0; kt < 8; ++kt)
        a[kt] = Xp[(mtile * 8 + kt) * 32 + lane];

    float dacc[8];
    #pragma unroll
    for (int v = 0; v < 8; ++v) dacc[v] = 0.0f;

    // ---- GEMM1: H1 = relu(X @ W1 + b1) -> LDS bf16 ----
    #pragma unroll
    for (int nt = 0; nt < 8; nt += 2) {
        v8f c0 = {}, c1 = {};
        v16bf bA = W1p[(0 * 8 + nt) * 32 + lane];
        v16bf bB = W1p[(0 * 8 + nt + 1) * 32 + lane];
        #pragma unroll
        for (int kt = 0; kt < 8; ++kt) {
            v16bf nA, nB;
            if (kt < 7) {
                nA = W1p[((kt + 1) * 8 + nt) * 32 + lane];
                nB = W1p[((kt + 1) * 8 + nt + 1) * 32 + lane];
            }
            c0 = __builtin_amdgcn_wmma_f32_16x16x32_bf16(
                     false, a[kt], false, bA, (short)0, c0, false, false);
            c1 = __builtin_amdgcn_wmma_f32_16x16x32_bf16(
                     false, a[kt], false, bB, (short)0, c1, false, false);
            if (kt < 7) { bA = nA; bB = nB; }
        }
        int col0 = nt * 16 + nl;
        float bias0 = b1[col0], bias1 = b1[col0 + 16];
        #pragma unroll
        for (int v = 0; v < 8; ++v) {
            float h0 = c0[v] + bias0; h0 = h0 > 0.0f ? h0 : 0.0f;
            float h1 = c1[v] + bias1; h1 = h1 > 0.0f ? h1 : 0.0f;
            lds_h1[v + half * 8][col0]      = (__bf16)h0;
            lds_h1[v + half * 8][col0 + 16] = (__bf16)h1;
        }
    }
    __syncthreads();

    // ---- GEMM2: H2 = relu(H1 @ W2 + b2), fused dot with Wo_deep ----
    v16bf a2[4];
    #pragma unroll
    for (int kt = 0; kt < 4; ++kt) {
        #pragma unroll
        for (int e = 0; e < 16; ++e) {
            int K = half * 16 + e;
            a2[kt][e] = lds_h1[nl][kt * 32 + K];
        }
    }
    #pragma unroll
    for (int nt = 0; nt < 8; nt += 2) {
        v8f c0 = {}, c1 = {};
        v16bf bA = W2p[(0 * 8 + nt) * 32 + lane];
        v16bf bB = W2p[(0 * 8 + nt + 1) * 32 + lane];
        #pragma unroll
        for (int kt = 0; kt < 4; ++kt) {
            v16bf nA, nB;
            if (kt < 3) {
                nA = W2p[((kt + 1) * 8 + nt) * 32 + lane];
                nB = W2p[((kt + 1) * 8 + nt + 1) * 32 + lane];
            }
            c0 = __builtin_amdgcn_wmma_f32_16x16x32_bf16(
                     false, a2[kt], false, bA, (short)0, c0, false, false);
            c1 = __builtin_amdgcn_wmma_f32_16x16x32_bf16(
                     false, a2[kt], false, bB, (short)0, c1, false, false);
            if (kt < 3) { bA = nA; bB = nB; }
        }
        int col0 = nt * 16 + nl;
        float bias0 = b2[col0], bias1 = b2[col0 + 16];
        float wod0 = Wo[N_PAIRS + col0], wod1 = Wo[N_PAIRS + col0 + 16];
        #pragma unroll
        for (int v = 0; v < 8; ++v) {
            float h0 = c0[v] + bias0; h0 = h0 > 0.0f ? h0 : 0.0f;
            float h1 = c1[v] + bias1; h1 = h1 > 0.0f ? h1 : 0.0f;
            dacc[v] += h0 * wod0 + h1 * wod1;
        }
    }

    // ---- GEMM3: Y = X @ M', fused wide dot with f32 x strip ----
    #pragma unroll
    for (int nt = 0; nt < 16; nt += 2) {
        v8f c0 = {}, c1 = {};
        v16bf bA = Mp[(0 * 16 + nt) * 32 + lane];
        v16bf bB = Mp[(0 * 16 + nt + 1) * 32 + lane];
        #pragma unroll
        for (int kt = 0; kt < 8; ++kt) {
            v16bf nA, nB;
            if (kt < 7) {
                nA = Mp[((kt + 1) * 16 + nt) * 32 + lane];
                nB = Mp[((kt + 1) * 16 + nt + 1) * 32 + lane];
            }
            c0 = __builtin_amdgcn_wmma_f32_16x16x32_bf16(
                     false, a[kt], false, bA, (short)0, c0, false, false);
            c1 = __builtin_amdgcn_wmma_f32_16x16x32_bf16(
                     false, a[kt], false, bB, (short)0, c1, false, false);
            if (kt < 7) { bA = nA; bB = nB; }
        }
        int col0 = nt * 16 + nl;
        #pragma unroll
        for (int v = 0; v < 8; ++v) {
            int row = v + half * 8;
            dacc[v] += c0[v] * lds_x[row][col0]
                     + c1[v] * lds_x[row][col0 + 16];
        }
    }

    // ---- cross-lane reduction + sigmoid ----
    #pragma unroll
    for (int v = 0; v < 8; ++v) lds_red[lane][v] = dacc[v];
    __syncthreads();

    if (lane < 16) {
        int r = lane;
        float s = 0.0f;
        if (r < 8) {
            #pragma unroll
            for (int l = 0; l < 16; ++l) s += lds_red[l][r];
        } else {
            #pragma unroll
            for (int l = 0; l < 16; ++l) s += lds_red[16 + l][r - 8];
        }
        float z = s + bo[0];
        out[mtile * 16 + r] = 1.0f / (1.0f + expf(-z));
    }
}

// ---------------------------------------------------------------------------
extern "C" void kernel_launch(void* const* d_in, const int* in_sizes, int n_in,
                              void* d_out, int out_size, void* d_ws, size_t ws_size,
                              hipStream_t stream) {
    const float* x      = (const float*)d_in[0];
    const float* w_wide = (const float*)d_in[1];
    const float* W1     = (const float*)d_in[2];
    const float* b1     = (const float*)d_in[3];
    const float* W2     = (const float*)d_in[4];
    const float* b2     = (const float*)d_in[5];
    const float* Wo     = (const float*)d_in[6];
    const float* bo     = (const float*)d_in[7];
    float* out = (float*)d_out;

    char* ws = (char*)d_ws;
    __bf16* Mp  = (__bf16*)(ws + 0);          // 131072 B
    __bf16* W1p = (__bf16*)(ws + 131072);     //  65536 B
    __bf16* W2p = (__bf16*)(ws + 196608);     //  32768 B
    __bf16* Xp  = (__bf16*)(ws + 229376);     // 2 MB

    pack_M_kernel<<<65536 / 256, 256, 0, stream>>>(w_wide, Wo, Mp);
    pack_W_kernel<<<(N_FEAT * HIDDEN) / 256, 256, 0, stream>>>(W1, W1p, N_FEAT, HIDDEN);
    pack_W_kernel<<<(HIDDEN * HIDDEN) / 256, 256, 0, stream>>>(W2, W2p, HIDDEN, HIDDEN);
    pack_X_kernel<<<(BATCH * N_FEAT) / 256, 256, 0, stream>>>(x, Xp);

    wd_main_kernel<<<BATCH / 16, 32, 0, stream>>>(
        x, b1, b2, Wo, bo,
        (const v16bf*)Xp, (const v16bf*)Mp, (const v16bf*)W1p, (const v16bf*)W2p,
        out);
}
